// Attention_10007273800154
// MI455X (gfx1250) — compile-verified, matrix operand order checked by reading, not run on hardware
//
#include <hip/hip_runtime.h>

// ---------------------------------------------------------------------------
// Problem constants (from reference): N1=N2=ITEM=4096, DIM=1000 (pad to 1024)
// ---------------------------------------------------------------------------
#define N1G   4096
#define N2G   4096
#define ITEMG 4096
#define DIMP  1024
#define DIMV  1000

typedef __attribute__((ext_vector_type(16))) __bf16 v16bf;
typedef __attribute__((ext_vector_type(8)))  float  v8f;
typedef __attribute__((ext_vector_type(4)))  float  f4v;
typedef __attribute__((ext_vector_type(8)))  __bf16 bf8v;

__device__ __forceinline__ v8f zero8() {
    v8f z;
#pragma unroll
    for (int i = 0; i < 8; ++i) z[i] = 0.0f;
    return z;
}

// Load one 16x32 bf16 WMMA operand fragment for NT layout (row-major over K).
// Per CDNA5 ISA 7.12.2: lane<16 holds row (base+lane), K = {0..7, 16..23};
// lane>=16 holds row (base+lane-16), K = {8..15, 24..31}; packed 2/VGPR.
// => two contiguous 16-byte chunks per lane (global_load_b128 x2).
__device__ __forceinline__ v16bf load_frag(const __bf16* __restrict__ base,
                                           int ld, int row0, int k0) {
    const int lane = threadIdx.x & 31;
    const int l = lane & 15;
    const int hi = lane >> 4;
    const __bf16* p = base + (size_t)(row0 + l) * (size_t)ld + (size_t)(k0 + hi * 8);
    union { v16bf v; f4v f[2]; } u;
    u.f[0] = *(const f4v*)p;         // K chunk 0 (8 bf16 = 16B)
    u.f[1] = *(const f4v*)(p + 16);  // K chunk 1 (+16 elements)
    return u.v;
}

// Core: one wave computes a 64x64 f32 tile of C = A * B^T (both row-major
// over K, K % 32 == 0) using 16 v_wmma_f32_16x16x32_bf16 per K-step.
// Unconditional next-step prefetch: speculative global_prefetch_b8 past the
// region end is silently dropped (ISA 10.5), and workspace is contiguous.
__device__ __forceinline__ void gemm64_nt(const __bf16* __restrict__ A, int lda,
                                          const __bf16* __restrict__ B, int ldb,
                                          int m0, int n0, int K, v8f acc[4][4]) {
#pragma unroll
    for (int i = 0; i < 4; ++i)
#pragma unroll
        for (int j = 0; j < 4; ++j) acc[i][j] = zero8();

    const int lane = threadIdx.x & 31;
    for (int k0 = 0; k0 < K; k0 += 32) {
        // prefetch next K-step cachelines (one 64B line covers K-range of 32)
        __builtin_prefetch(A + (size_t)(m0 + lane * 2) * (size_t)lda + (k0 + 32), 0, 3);
        __builtin_prefetch(B + (size_t)(n0 + lane * 2) * (size_t)ldb + (k0 + 32), 0, 3);

        v16bf a[4], b[4];
#pragma unroll
        for (int i = 0; i < 4; ++i) a[i] = load_frag(A, lda, m0 + 16 * i, k0);
#pragma unroll
        for (int j = 0; j < 4; ++j) b[j] = load_frag(B, ldb, n0 + 16 * j, k0);
#pragma unroll
        for (int i = 0; i < 4; ++i)
#pragma unroll
            for (int j = 0; j < 4; ++j)
                acc[i][j] = __builtin_amdgcn_wmma_f32_16x16x32_bf16(
                    false, a[i], false, b[j], (short)0, acc[i][j], false, false);
    }
}

// ---------------------------------------------------------------------------
// Conversion / layout kernels
// ---------------------------------------------------------------------------
__global__ void cvt_bf16_kernel(const float* __restrict__ in,
                                __bf16* __restrict__ out, size_t n) {
    size_t idx = ((size_t)blockIdx.x * blockDim.x + threadIdx.x) * 8;
    if (idx >= n) return;
    f4v a = *(const f4v*)(in + idx);
    f4v b = *(const f4v*)(in + idx + 4);
    bf8v o;
#pragma unroll
    for (int k = 0; k < 4; ++k) { o[k] = (__bf16)a[k]; o[k + 4] = (__bf16)b[k]; }
    *(bf8v*)(out + idx) = o;
}

// Wq/Wk [1000,4096] f32 -> [1024,4096] bf16 with rows >= 1000 zeroed.
__global__ void cvt_w_kernel(const float* __restrict__ in, __bf16* __restrict__ out) {
    size_t idx = ((size_t)blockIdx.x * blockDim.x + threadIdx.x) * 8;
    if (idx >= (size_t)DIMP * ITEMG) return;
    int row = (int)(idx / ITEMG);
    bf8v o;
    if (row < DIMV) {
        f4v a = *(const f4v*)(in + idx);      // same row-major layout for valid rows
        f4v b = *(const f4v*)(in + idx + 4);
#pragma unroll
        for (int k = 0; k < 4; ++k) { o[k] = (__bf16)a[k]; o[k + 4] = (__bf16)b[k]; }
    } else {
#pragma unroll
        for (int k = 0; k < 8; ++k) o[k] = (__bf16)0.0f;
    }
    *(bf8v*)(out + idx) = o;
}

// x2 [4096,4096] f32 -> x2^T [4096,4096] bf16 (LDS tile transpose, no bank conflicts)
__global__ void transpose_cvt_kernel(const float* __restrict__ in,
                                     __bf16* __restrict__ out, int dim) {
    __shared__ __bf16 tile[32][33];
    const int bx = blockIdx.x * 32, by = blockIdx.y * 32;
    const int tx = threadIdx.x, ty = threadIdx.y;  // 32 x 8
#pragma unroll
    for (int i = 0; i < 32; i += 8)
        tile[ty + i][tx] = (__bf16)in[(size_t)(by + ty + i) * dim + (bx + tx)];
    __syncthreads();
#pragma unroll
    for (int i = 0; i < 32; i += 8)
        out[(size_t)(bx + ty + i) * dim + (by + tx)] = tile[tx][ty + i];
}

// ---------------------------------------------------------------------------
// GEMM kernels (128 threads = 4 waves, 2x2 arrangement -> 128x128 block tile)
// ---------------------------------------------------------------------------
// Q = x1 @ Wq^T + bq  (and K = x2 @ Wk^T + bk), output bf16 [rows, 1024],
// columns >= 1000 stored as zero so the padded scores-GEMM K-dim is exact.
__global__ void __launch_bounds__(128)
proj_kernel(const __bf16* __restrict__ X, const __bf16* __restrict__ W,
            const float* __restrict__ bias, __bf16* __restrict__ Out) {
    const int wave = threadIdx.x >> 5;
    const int m0 = blockIdx.y * 128 + (wave >> 1) * 64;
    const int n0 = blockIdx.x * 128 + (wave & 1) * 64;
    v8f acc[4][4];
    gemm64_nt(X, ITEMG, W, ITEMG, m0, n0, ITEMG, acc);
    const int lane = threadIdx.x & 31, l = lane & 15, hi = lane >> 4;
#pragma unroll
    for (int i = 0; i < 4; ++i)
#pragma unroll
        for (int j = 0; j < 4; ++j) {
            const int n = n0 + 16 * j + l;
            const float bn = (n < DIMV) ? bias[n] : 0.0f;
#pragma unroll
            for (int r = 0; r < 8; ++r) {
                int m = m0 + 16 * i + r + 8 * hi;
                float v = (n < DIMV) ? (acc[i][j][r] + bn) : 0.0f;
                Out[(size_t)m * DIMP + n] = (__bf16)v;
            }
        }
}

// e = exp((Q @ K^T)/64) * label_map, stored bf16.
__global__ void __launch_bounds__(128)
scores_kernel(const __bf16* __restrict__ Q, const __bf16* __restrict__ Kb,
              const float* __restrict__ label, __bf16* __restrict__ E) {
    const int wave = threadIdx.x >> 5;
    const int m0 = blockIdx.y * 128 + (wave >> 1) * 64;
    const int n0 = blockIdx.x * 128 + (wave & 1) * 64;
    v8f acc[4][4];
    gemm64_nt(Q, DIMP, Kb, DIMP, m0, n0, DIMP, acc);
    const int lane = threadIdx.x & 31, l = lane & 15, hi = lane >> 4;
    const float inv_denom = 0.015625f;  // 1/sqrt(4096)
#pragma unroll
    for (int i = 0; i < 4; ++i)
#pragma unroll
        for (int j = 0; j < 4; ++j)
#pragma unroll
            for (int r = 0; r < 8; ++r) {
                int m = m0 + 16 * i + r + 8 * hi;
                int n = n0 + 16 * j + l;
                float sc = acc[i][j][r] * inv_denom;
                float ev = __expf(sc) * label[(size_t)m * N2G + n];
                E[(size_t)m * N2G + n] = (__bf16)ev;
            }
}

// Per-row: s[m] = (sum(label)/topk + 1e-8) / (sum(e) + 1e-8)
__global__ void row_stats_kernel(const __bf16* __restrict__ E,
                                 const float* __restrict__ label,
                                 const int* __restrict__ topk,
                                 float* __restrict__ S) {
    const int m = blockIdx.x;
    const int tid = threadIdx.x;
    float esum = 0.0f, lsum = 0.0f;
    for (int c = tid; c < N2G; c += blockDim.x) {
        esum += (float)E[(size_t)m * N2G + c];
        lsum += label[(size_t)m * N2G + c];
    }
    __shared__ float se[256], sl[256];
    se[tid] = esum; sl[tid] = lsum;
    __syncthreads();
    for (int s2 = 128; s2 > 0; s2 >>= 1) {
        if (tid < s2) { se[tid] += se[tid + s2]; sl[tid] += sl[tid + s2]; }
        __syncthreads();
    }
    if (tid == 0)
        S[m] = (sl[0] / (float)topk[0] + 1e-8f) / (se[0] + 1e-8f);
}

// out = (e @ x2) * s[m], computed as NT GEMM against x2^T, f32 output.
__global__ void __launch_bounds__(128)
out_kernel(const __bf16* __restrict__ E, const __bf16* __restrict__ X2T,
           const float* __restrict__ S, float* __restrict__ Out) {
    const int wave = threadIdx.x >> 5;
    const int m0 = blockIdx.y * 128 + (wave >> 1) * 64;
    const int n0 = blockIdx.x * 128 + (wave & 1) * 64;
    v8f acc[4][4];
    gemm64_nt(E, N2G, X2T, N2G, m0, n0, N2G, acc);
    const int lane = threadIdx.x & 31, l = lane & 15, hi = lane >> 4;
#pragma unroll
    for (int i = 0; i < 4; ++i) {
        float sm[8];
#pragma unroll
        for (int r = 0; r < 8; ++r) sm[r] = S[m0 + 16 * i + r + 8 * hi];
#pragma unroll
        for (int j = 0; j < 4; ++j)
#pragma unroll
            for (int r = 0; r < 8; ++r) {
                int m = m0 + 16 * i + r + 8 * hi;
                int n = n0 + 16 * j + l;
                Out[(size_t)m * ITEMG + n] = acc[i][j][r] * sm[r];
            }
    }
}

// ---------------------------------------------------------------------------
// Host launcher
// ---------------------------------------------------------------------------
extern "C" void kernel_launch(void* const* d_in, const int* in_sizes, int n_in,
                              void* d_out, int out_size, void* d_ws, size_t ws_size,
                              hipStream_t stream) {
    const float* x1    = (const float*)d_in[0];
    const float* x2    = (const float*)d_in[1];
    const float* label = (const float*)d_in[2];
    const float* Wq    = (const float*)d_in[3];
    const float* bq    = (const float*)d_in[4];
    const float* Wk    = (const float*)d_in[5];
    const float* bk    = (const float*)d_in[6];
    const int*   topk  = (const int*)d_in[7];
    float* out = (float*)d_out;

    // workspace carve-out (~168 MB)
    char* ws = (char*)d_ws;
    __bf16* x1b  = (__bf16*)ws; ws += (size_t)N1G * ITEMG * 2;
    __bf16* x2b  = (__bf16*)ws; ws += (size_t)N2G * ITEMG * 2;
    __bf16* x2tb = (__bf16*)ws; ws += (size_t)ITEMG * N2G * 2;
    __bf16* Wqb  = (__bf16*)ws; ws += (size_t)DIMP * ITEMG * 2;
    __bf16* Wkb  = (__bf16*)ws; ws += (size_t)DIMP * ITEMG * 2;
    __bf16* Qb   = (__bf16*)ws; ws += (size_t)N1G * DIMP * 2;
    __bf16* Kb   = (__bf16*)ws; ws += (size_t)N2G * DIMP * 2;
    __bf16* Eb   = (__bf16*)ws; ws += (size_t)N1G * N2G * 2;
    float*  Srow = (float*)ws;  ws += (size_t)N1G * 4;

    const size_t nx = (size_t)N1G * ITEMG;

    // 1) precision / layout conversions
    cvt_bf16_kernel<<<(unsigned)(nx / 8 / 256), 256, 0, stream>>>(x1, x1b, nx);
    cvt_bf16_kernel<<<(unsigned)(nx / 8 / 256), 256, 0, stream>>>(x2, x2b, nx);
    cvt_w_kernel<<<(unsigned)((size_t)DIMP * ITEMG / 8 / 256), 256, 0, stream>>>(Wq, Wqb);
    cvt_w_kernel<<<(unsigned)((size_t)DIMP * ITEMG / 8 / 256), 256, 0, stream>>>(Wk, Wkb);
    transpose_cvt_kernel<<<dim3(N2G / 32, ITEMG / 32), dim3(32, 8), 0, stream>>>(x2, x2tb, N2G);

    // 2) Q/K projections (WMMA)
    proj_kernel<<<dim3(DIMP / 128, N1G / 128), 128, 0, stream>>>(x1b, Wqb, bq, Qb);
    proj_kernel<<<dim3(DIMP / 128, N2G / 128), 128, 0, stream>>>(x2b, Wkb, bk, Kb);

    // 3) masked-exp scores (WMMA + exp epilogue)
    scores_kernel<<<dim3(N2G / 128, N1G / 128), 128, 0, stream>>>(Qb, Kb, label, Eb);

    // 4) per-row normalization factors
    row_stats_kernel<<<N1G, 256, 0, stream>>>(Eb, label, topk, Srow);

    // 5) final aggregation att @ x2 (WMMA, f32 out)
    out_kernel<<<dim3(ITEMG / 128, N1G / 128), 128, 0, stream>>>(Eb, x2tb, Srow, out);
}
